// GraphPositionalEncoding_68667937129202
// MI455X (gfx1250) — matrix operand order repaired and checked by reading, not compiled, not run on hardware
//
#include <hip/hip_runtime.h>

#define HIDDEN   256
#define NODES    256
#define KDIM     8
#define EPG      2048
#define NGRAPH   256
#define NTHREADS 1024

typedef __attribute__((ext_vector_type(16))) __bf16 v16bf;
typedef __attribute__((ext_vector_type(8)))  float  v8f;

struct Frag32 { uint4 q0, q1; };   // 32 bytes = one 16x32 bf16 fragment per lane

__device__ __forceinline__ v16bf frag_bf(const Frag32& f) {
    return __builtin_bit_cast(v16bf, f);
}

__device__ __forceinline__ unsigned short f2bf(float f) {
    unsigned int u = __float_as_uint(f);
    u += 0x7FFFu + ((u >> 16) & 1u);     // round-to-nearest-even
    return (unsigned short)(u >> 16);
}
__device__ __forceinline__ float bf2f(unsigned short h) {
    return __uint_as_float(((unsigned int)h) << 16);
}

__global__ void zero_deg_kernel(int* __restrict__ deg, int n) {
    int t = blockIdx.x * blockDim.x + threadIdx.x;
    if (t < n) deg[t] = 0;
}

__global__ void scatter_deg_kernel(const int* __restrict__ dst, int* __restrict__ deg, int ne) {
    int t = blockIdx.x * blockDim.x + threadIdx.x;
    if (t < ne) atomicAdd(&deg[dst[t]], 1);
}

// One workgroup per graph. 32 waves. LDS-resident bf16 T^T and P matrices.
__global__ __launch_bounds__(NTHREADS, 1)
void rwpe_kernel(const int*   __restrict__ ei_src,
                 const int*   __restrict__ ei_dst,
                 const int*   __restrict__ deg,
                 const float* __restrict__ degree_table,
                 const float* __restrict__ rw_w,
                 const float* __restrict__ rw_b,
                 float*       __restrict__ out)
{
    extern __shared__ char smem[];
    unsigned short* Tm   = (unsigned short*)smem;        // [256][256] bf16 : T^T row-major (== T col-major)
    unsigned short* Pm   = Tm + NODES * NODES;           // [256][256] bf16 : power matrix, row-major
    float*          diag = (float*)(Pm + NODES * NODES); // [KDIM][256]
    float*          dinv = diag + KDIM * NODES;          // [256]
    int*            eLds = (int*)(dinv + NODES);         // [2][2048] staged edge slice (16 KB)

    const int g    = blockIdx.x;
    const int t    = threadIdx.x;
    const int lane = t & 31;
    const int wave = t >> 5;
    const int hi   = (lane >= 16) ? 1 : 0;
    const int lm   = lane & 15;
    const int ebase = g * EPG;

    // ---- async-stage this graph's edge slice into LDS (overlapped with zero-init) ----
    {
        const unsigned int eOff = (unsigned int)(uintptr_t)eLds;  // LDS byte offset (addr[31:0])
        if (wave < 16) {
            unsigned int lo = eOff + (unsigned int)t * 16u;
            unsigned int go = (unsigned int)t * 16u;
            asm volatile("global_load_async_to_lds_b128 %0, %1, %2"
                         :: "v"(lo), "v"(go), "s"(ei_src + ebase) : "memory");
        } else {
            unsigned int lo = eOff + 8192u + (unsigned int)(t - 512) * 16u;
            unsigned int go = (unsigned int)(t - 512) * 16u;
            asm volatile("global_load_async_to_lds_b128 %0, %1, %2"
                         :: "v"(lo), "v"(go), "s"(ei_dst + ebase) : "memory");
        }
    }

    // ---- init: adjacency = 0, P = I (overlaps with async edge DMA) ----
    for (int e = t; e < NODES * NODES; e += NTHREADS) {
        Tm[e] = 0;
        Pm[e] = ((e >> 8) == (e & 255)) ? (unsigned short)0x3F80u : (unsigned short)0u;
    }
    asm volatile("s_wait_asynccnt 0x0" ::: "memory");
    __syncthreads();

    // ---- scatter edges from LDS, symmetrized (benign same-value write races) ----
    for (int e = t; e < EPG; e += NTHREADS) {
        int s = eLds[e] & (NODES - 1);
        int d = eLds[EPG + e] & (NODES - 1);
        Tm[s * NODES + d] = (unsigned short)0x3F80u;
        Tm[d * NODES + s] = (unsigned short)0x3F80u;
    }
    __syncthreads();

    // ---- row sums of symmetric adjacency -> dinv (counts <= 256 are bf16-exact) ----
    if (t < NODES) {
        float s = 0.f;
        const unsigned short* row = Tm + t * NODES;
        for (int j = 0; j < NODES; ++j) s += bf2f(row[j]);
        dinv[t] = (s > 0.f) ? (1.0f / s) : 0.f;
    }
    __syncthreads();

    // ---- T^T[q][i] = adj[q][i] * dinv[q]  (row scaling of symmetric adj == col scaling of T) ----
    for (int e = t; e < NODES * NODES; e += NTHREADS) {
        int q = e >> 8;
        Tm[e] = f2bf(bf2f(Tm[e]) * dinv[q]);
    }
    __syncthreads();

    // ---- 8x (P = P @ T) with diag harvest; WMMA bf16 -> f32 ----
    const int r  = wave >> 1;            // row tile 0..15 (16 rows each)
    const int ch = wave & 1;             // column half (8 tiles of 16 cols)
    const int m  = (r << 4) + lm;        // A-matrix row this lane supplies

    const char* aRowBase = (const char*)Pm + m * (NODES * 2);
    // All 8 B tiles for this wave share one base; tiles are 8 KB apart (fits DS imm offsets)
    const char* bColBase = (const char*)Tm + ((ch << 3) * 16 + lm) * (NODES * 2);

    for (int step = 0; step < KDIM; ++step) {
        v8f acc[8];
        #pragma unroll
        for (int n = 0; n < 8; ++n)
            #pragma unroll
            for (int d = 0; d < 8; ++d) acc[n][d] = 0.f;

        // One-deep software pipeline over the 64 (kt, n) WMMAs: issue fragment i+1's
        // ds_load_b128s, then WMMA i, so each s_wait_dscnt has a WMMA of slack.
        Frag32 A2[2], B2[2];
        {   // prologue: fragment 0
            const char* ab = aRowBase + 0 * 64 + hi * 16;
            A2[0].q0 = *(const uint4*)(ab);
            A2[0].q1 = *(const uint4*)(ab + 32);
            const char* bb = bColBase + 0 * 64 + hi * 16;
            B2[0].q0 = *(const uint4*)(bb);
            B2[0].q1 = *(const uint4*)(bb + 32);
        }
        #pragma unroll
        for (int idx = 0; idx < 64; ++idx) {
            const int kt = idx >> 3;
            const int n  = idx & 7;
            if (idx + 1 < 64) {
                const int ni  = idx + 1;
                const int nkt = ni >> 3;
                const int nn  = ni & 7;
                if (nn == 0) {
                    const char* ab = aRowBase + nkt * 64 + hi * 16;
                    A2[nkt & 1].q0 = *(const uint4*)(ab);
                    A2[nkt & 1].q1 = *(const uint4*)(ab + 32);
                }
                const char* bb = bColBase + nn * (16 * NODES * 2) + nkt * 64 + hi * 16;
                B2[ni & 1].q0 = *(const uint4*)(bb);
                B2[ni & 1].q1 = *(const uint4*)(bb + 32);
            }
            acc[n] = __builtin_amdgcn_wmma_f32_16x16x32_bf16(
                false, frag_bf(A2[kt & 1]), false, frag_bf(B2[idx & 1]),
                (short)0, acc[n], false, false);
        }
        __syncthreads();   // all reads of P complete before overwrite

        #pragma unroll
        for (int n = 0; n < 8; ++n) {
            const int nt  = (ch << 3) + n;
            const int col = (nt << 4) + lm;
            #pragma unroll
            for (int d = 0; d < 8; ++d) {
                const int row = (r << 4) + d + (hi ? 8 : 0);
                const float v = acc[n][d];
                Pm[row * NODES + col] = f2bf(v);
                if (nt == r && row == col) diag[step * NODES + row] = v;  // diag(T^(step+1))
            }
        }
        __syncthreads();
    }

    // ---- epilogue: out = degree_table[deg] + diag @ rw_w^T + rw_b ----
    const int i  = t >> 2;               // node within graph
    const int h0 = (t & 3) << 6;         // 64 hidden columns per thread
    float dv[KDIM];
    #pragma unroll
    for (int k = 0; k < KDIM; ++k) dv[k] = diag[k * NODES + i];

    const int node = g * NODES + i;
    int dg = deg[node];
    if (dg > NODES - 1) dg = NODES - 1;
    const float* pe_row = degree_table + dg * HIDDEN;
    __builtin_prefetch(pe_row + h0, 0, 1);      // global_prefetch_b8
    float* orow = out + (size_t)node * HIDDEN;
    for (int h = h0; h < h0 + 64; ++h) {
        float a = rw_b[h] + pe_row[h];
        #pragma unroll
        for (int k = 0; k < KDIM; ++k) a += dv[k] * rw_w[h * KDIM + k];
        orow[h] = a;
    }
}

extern "C" void kernel_launch(void* const* d_in, const int* in_sizes, int n_in,
                              void* d_out, int out_size, void* d_ws, size_t ws_size,
                              hipStream_t stream) {
    (void)in_sizes; (void)n_in; (void)out_size; (void)ws_size;
    // setup_inputs order: 0=x, 1=edge_index[2][B*E], 2=batch, 3=degree_table, 4=rw_w, 5=rw_b
    const int*   ei   = (const int*)d_in[1];
    const int*   src  = ei;
    const int*   dst  = ei + NGRAPH * EPG;
    const float* dtab = (const float*)d_in[3];
    const float* rww  = (const float*)d_in[4];
    const float* rwb  = (const float*)d_in[5];
    float*       out  = (float*)d_out;

    int* deg = (int*)d_ws;               // 65536 ints of scratch
    const int num_nodes = NGRAPH * NODES;
    const int num_edges = NGRAPH * EPG;

    zero_deg_kernel<<<(num_nodes + 255) / 256, 256, 0, stream>>>(deg, num_nodes);
    scatter_deg_kernel<<<(num_edges + 255) / 256, 256, 0, stream>>>(dst, deg, num_edges);

    const size_t shmem = (size_t)(2 * NODES * NODES) * sizeof(unsigned short)  // Tm + Pm
                       + (size_t)(KDIM * NODES + NODES) * sizeof(float)        // diag + dinv
                       + (size_t)(2 * EPG) * sizeof(int);                      // staged edges
    rwpe_kernel<<<NGRAPH, NTHREADS, shmem, stream>>>(src, dst, deg, dtab, rww, rwb, out);
}
